// RingAttentionLlama_2602750181654
// MI455X (gfx1250) — matrix operand order repaired and checked by reading, not compiled, not run on hardware
//
#include <hip/hip_runtime.h>

// ---------------- problem constants (match reference) ----------------
static constexpr int Bc   = 2;
static constexpr int S    = 2048;
static constexpr int D    = 4096;
static constexpr int H    = 32;
static constexpr int KVH  = 8;
static constexpr int HD   = 128;
static constexpr int NREP = H / KVH;        // 4
static constexpr int E    = H * HD;         // 4096
static constexpr int EKV  = KVH * HD;       // 1024
static constexpr int M    = Bc * S;         // 4096 rows
static constexpr float SCALE = 0.08838834764831845f;   // HD^-0.5
static constexpr float NEG   = -1.0e9f;

// ---------------- CDNA5 vector types ----------------
typedef __attribute__((ext_vector_type(16))) __bf16       v16bf;
typedef __attribute__((ext_vector_type(8)))  float        v8f;
typedef __attribute__((ext_vector_type(4)))  unsigned int u32x4;

union FragAB { v16bf v; u32x4 u[2]; };

// ---------------- CDNA5 async global->LDS path (guarded) ----------------
#ifndef __has_builtin
#define __has_builtin(x) 0
#endif
#if __has_builtin(__builtin_amdgcn_global_load_async_to_lds_b128) && \
    __has_builtin(__builtin_amdgcn_s_wait_asynccnt)
#define USE_ASYNC_LDS 1
#else
#define USE_ASYNC_LDS 0
#endif

// exact pointee type the builtin expects: int __attribute__((vector_size(16)))
typedef int v4i __attribute__((vector_size(16)));
typedef __attribute__((address_space(1))) v4i* as1_v4i_p;   // global src
typedef __attribute__((address_space(3))) v4i* as3_v4i_p;   // LDS dst

__device__ __forceinline__ void async_copy_b128x2(const __bf16* g, __bf16* l) {
#if USE_ASYNC_LDS
    // 2 x b128 = 16 contiguous bf16, tracked by ASYNCcnt (no VGPR data path)
    as1_v4i_p gp = (as1_v4i_p)g;
    as3_v4i_p lp = (as3_v4i_p)l;
    __builtin_amdgcn_global_load_async_to_lds_b128(gp, lp, 0, 0);
    __builtin_amdgcn_global_load_async_to_lds_b128(gp, lp, 16, 0);
#else
    u32x4 a0 = *(const u32x4*)(g);
    u32x4 a1 = *(const u32x4*)(g + 8);
    *(u32x4*)(l)     = a0;
    *(u32x4*)(l + 8) = a1;
#endif
}
__device__ __forceinline__ void async_copy_b128x4(const __bf16* g, __bf16* l) {
#if USE_ASYNC_LDS
    as1_v4i_p gp = (as1_v4i_p)g;
    as3_v4i_p lp = (as3_v4i_p)l;
    __builtin_amdgcn_global_load_async_to_lds_b128(gp, lp, 0, 0);
    __builtin_amdgcn_global_load_async_to_lds_b128(gp, lp, 16, 0);
    __builtin_amdgcn_global_load_async_to_lds_b128(gp, lp, 32, 0);
    __builtin_amdgcn_global_load_async_to_lds_b128(gp, lp, 48, 0);
#else
    u32x4 a0 = *(const u32x4*)(g);
    u32x4 a1 = *(const u32x4*)(g + 8);
    u32x4 a2 = *(const u32x4*)(g + 16);
    u32x4 a3 = *(const u32x4*)(g + 24);
    *(u32x4*)(l)      = a0;
    *(u32x4*)(l + 8)  = a1;
    *(u32x4*)(l + 16) = a2;
    *(u32x4*)(l + 24) = a3;
#endif
}
__device__ __forceinline__ void async_wait0() {
#if USE_ASYNC_LDS
    __builtin_amdgcn_s_wait_asynccnt(0);
#endif
}

__device__ __forceinline__ __bf16 f2bf(float f) {
    unsigned u = __builtin_bit_cast(unsigned, f);
    unsigned r = (u + 0x7FFFu + ((u >> 16) & 1u)) >> 16;   // round-nearest-even
    unsigned short h = (unsigned short)r;
    return __builtin_bit_cast(__bf16, h);
}

// ---------------- fp32 -> bf16 conversion ----------------
__global__ void cvt_f32_to_bf16(const float* __restrict__ src,
                                __bf16* __restrict__ dst, long long n) {
    long long i = (long long)blockIdx.x * blockDim.x + threadIdx.x;
    long long stride = (long long)gridDim.x * blockDim.x;
    for (; i < n; i += stride) dst[i] = f2bf(src[i]);
}

// ---------------- LDS-staged double-buffered WMMA GEMM ----------------
// C[M,N] = A[M,K] * B[N,K]^T, bf16 in / f32 out.
// Block = 4 waves. Block tile 64(M) x 128(N), K-step 32.
// A/B tiles staged via GLOBAL_LOAD_ASYNC_TO_LDS (ASYNCcnt), double buffered so
// the async DMA of tile k+1 overlaps the ds_load + v_wmma of tile k.
__global__ __launch_bounds__(128)
void wmma_gemm_bf16(const __bf16* __restrict__ A, const __bf16* __restrict__ Bw,
                    float* __restrict__ C, int Mr, int Nr, int Kr) {
    __shared__ __align__(16) __bf16 sA[2][64 * 32];    // 8 KB
    __shared__ __align__(16) __bf16 sB[2][128 * 32];   // 16 KB
    const int tid  = threadIdx.x;
    const int lane = tid & 31;
    const int wave = tid >> 5;
    const int half = lane >> 4;
    const int lr   = lane & 15;
    const int m_blk = blockIdx.y * 64;
    const int n0    = blockIdx.x * 128;

    // cooperative copy mapping: A row = tid/2, 16-elem segment; B row = tid, 32 elems
    const int arow = tid >> 1, aseg = (tid & 1) * 16;
    const __bf16* gA = A  + (size_t)(m_blk + arow) * Kr + aseg;
    const __bf16* gB = Bw + (size_t)(n0 + tid) * Kr;

    v8f acc[8];
#pragma unroll
    for (int t = 0; t < 8; ++t) acc[t] = (v8f){0,0,0,0,0,0,0,0};

    const int nk = Kr / 32;
    // prologue: stage k-tile 0 into buffer 0
    async_copy_b128x2(gA, &sA[0][arow * 32 + aseg]);
    async_copy_b128x4(gB, &sB[0][tid * 32]);
    async_wait0();
    __syncthreads();

    for (int it = 0; it < nk; ++it) {
        const int cur = it & 1;
        if (it + 1 < nk) {   // kick off DMA for the next k-tile into the other buffer
            async_copy_b128x2(gA + (it + 1) * 32, &sA[cur ^ 1][arow * 32 + aseg]);
            async_copy_b128x4(gB + (it + 1) * 32, &sB[cur ^ 1][tid * 32]);
        }
        // A fragment from LDS (row = wave*16 + lr)
        FragAB a;
        const __bf16* la = &sA[cur][(wave * 16 + lr) * 32];
        a.u[0] = *(const u32x4*)(la + half * 8);
        a.u[1] = *(const u32x4*)(la + 16 + half * 8);
        // 8 B fragments, one per 16-wide N tile; A-frag reused 8x
#pragma unroll
        for (int t = 0; t < 8; ++t) {
            FragAB b;
            const __bf16* lb = &sB[cur][(t * 16 + lr) * 32 + half * 8];
            b.u[0] = *(const u32x4*)(lb);
            b.u[1] = *(const u32x4*)(lb + 16);
            acc[t] = __builtin_amdgcn_wmma_f32_16x16x32_bf16(false, a.v, false, b.v, (short)0, acc[t], false, false);
        }
        async_wait0();       // next buffer resident before it becomes 'cur'
        __syncthreads();     // all waves done reading 'cur' before it is overwritten
    }

#pragma unroll
    for (int r = 0; r < 8; ++r) {
        int row = m_blk + wave * 16 + r + 8 * half;
        float* crow = C + (size_t)row * Nr + n0 + lr;
#pragma unroll
        for (int t = 0; t < 8; ++t) crow[t * 16] = acc[t][r];
    }
}

// ---------------- RoPE + head-major repack: [B,S,nh,HD]f32 -> [B,nh,S,HD]bf16 ----------------
__global__ void rope_to_bf16(const float* __restrict__ src, const float* __restrict__ cs,
                             const float* __restrict__ sn, __bf16* __restrict__ dst, int nh) {
    long long total = (long long)Bc * S * nh * (HD / 2);
    long long gid = (long long)blockIdx.x * blockDim.x + threadIdx.x;
    long long stride = (long long)gridDim.x * blockDim.x;
    for (; gid < total; gid += stride) {
        int p = (int)(gid % (HD / 2));
        long long t = gid / (HD / 2);
        int h = (int)(t % nh); t /= nh;
        int s = (int)(t % S);
        int b = (int)(t / S);
        const float* base = src + (((size_t)(b * S + s) * nh + h) * HD);
        float t0 = base[2 * p], t1 = base[2 * p + 1];
        float c  = cs[(size_t)(b * S + s) * (HD / 2) + p];
        float si = sn[(size_t)(b * S + s) * (HD / 2) + p];
        __bf16* ob = dst + ((((size_t)b * nh + h) * S + s) * HD);
        ob[2 * p]     = f2bf(t0 * c - t1 * si);
        ob[2 * p + 1] = f2bf(t0 * si + t1 * c);
    }
}

// ---------------- V repack transposed: [B,S,KVH,HD]f32 -> [B,KVH,HD,S]bf16 ----------------
__global__ void vtrans_to_bf16(const float* __restrict__ src, __bf16* __restrict__ dst) {
    long long total = (long long)Bc * S * KVH * HD;
    long long gid = (long long)blockIdx.x * blockDim.x + threadIdx.x;
    long long stride = (long long)gridDim.x * blockDim.x;
    for (; gid < total; gid += stride) {
        int d = (int)(gid % HD);
        long long t = gid / HD;
        int kv = (int)(t % KVH); t /= KVH;
        int s = (int)(t % S);
        int b = (int)(t / S);
        dst[(((size_t)b * KVH + kv) * HD + d) * S + s] = f2bf(src[gid]);
    }
}

// ---------------- flash attention: one wave per (b, h, 16-query tile) ----------------
// 32-key steps: two score tiles per step, so the P*V WMMA uses the full K=32
// inner dim and every lane loads real V data (no zero padding).
__global__ __launch_bounds__(32)
void flash_attn(const __bf16* __restrict__ Q, const __bf16* __restrict__ Kc,
                const __bf16* __restrict__ Vt, const unsigned char* __restrict__ maskp,
                __bf16* __restrict__ O) {
    const int lane = threadIdx.x & 31;
    const int half = lane >> 4;
    const int lr   = lane & 15;
    const int qt = blockIdx.x, h = blockIdx.y, b = blockIdx.z;
    const int kv = h / NREP;
    __shared__ __align__(16) __bf16 lds_p[16 * 32];   // P tile (16 q x 32 keys)

    // Q tile fragments (16 queries x 128 dims = 4 K-chunks of 32)
    const __bf16* qbase = Q + ((((size_t)b * H + h) * S) + (size_t)qt * 16) * HD + (size_t)lr * HD;
    FragAB qf[4];
#pragma unroll
    for (int kk = 0; kk < 4; ++kk) {
        qf[kk].u[0] = *(const u32x4*)(qbase + kk * 32 + half * 8);
        qf[kk].u[1] = *(const u32x4*)(qbase + kk * 32 + 16 + half * 8);
    }

    const v8f zf = {0,0,0,0,0,0,0,0};
    v8f acc[8];
#pragma unroll
    for (int t = 0; t < 8; ++t) acc[t] = zf;
    float m_i[8], l_i[8];
#pragma unroll
    for (int r = 0; r < 8; ++r) { m_i[r] = -3.0e38f; l_i[r] = 0.0f; }

    const __bf16* kbase = Kc + (((size_t)b * KVH + kv) * S) * HD;
    const __bf16* vbase = Vt + (((size_t)b * KVH + kv) * HD) * S;
    const int njs = (qt >> 1) + 1;    // ceil((qt+1)*16 / 32)

    for (int j = 0; j < njs; ++j) {
        // scores for 32 keys = two 16x16 tiles (8 WMMAs along HD)
        v8f sc0 = zf, sc1 = zf;
        const __bf16* krow0 = kbase + ((size_t)j * 32 + lr) * HD + half * 8;
        const __bf16* krow1 = krow0 + (size_t)16 * HD;
        if (j + 1 < njs) __builtin_prefetch(krow0 + (size_t)32 * HD, 0, 0);  // global_prefetch_b8
#pragma unroll
        for (int kk = 0; kk < 4; ++kk) {
            FragAB kf0, kf1;
            kf0.u[0] = *(const u32x4*)(krow0 + kk * 32);
            kf0.u[1] = *(const u32x4*)(krow0 + kk * 32 + 16);
            kf1.u[0] = *(const u32x4*)(krow1 + kk * 32);
            kf1.u[1] = *(const u32x4*)(krow1 + kk * 32 + 16);
            sc0 = __builtin_amdgcn_wmma_f32_16x16x32_bf16(false, qf[kk].v, false, kf0.v, (short)0, sc0, false, false);
            sc1 = __builtin_amdgcn_wmma_f32_16x16x32_bf16(false, qf[kk].v, false, kf1.v, (short)0, sc1, false, false);
        }

        const int colg0 = j * 32 + lr;
        const int colg1 = colg0 + 16;
        const bool mok0 = maskp[(size_t)b * S + colg0] != 0;
        const bool mok1 = maskp[(size_t)b * S + colg1] != 0;

        // online softmax per row (rows replicated across the 16 lanes of each half)
#pragma unroll
        for (int r = 0; r < 8; ++r) {
            int rowg = qt * 16 + r + 8 * half;
            float s0 = (mok0 && colg0 <= rowg) ? sc0[r] * SCALE : NEG;
            float s1 = (mok1 && colg1 <= rowg) ? sc1[r] * SCALE : NEG;
            float mx = fmaxf(s0, s1);
            mx = fmaxf(mx, __shfl_xor(mx, 1, 32));
            mx = fmaxf(mx, __shfl_xor(mx, 2, 32));
            mx = fmaxf(mx, __shfl_xor(mx, 4, 32));
            mx = fmaxf(mx, __shfl_xor(mx, 8, 32));
            float mnew  = fmaxf(m_i[r], mx);
            float alpha = __expf(m_i[r] - mnew);
            float p0    = __expf(s0 - mnew);
            float p1    = __expf(s1 - mnew);
            float rs = p0 + p1;
            rs += __shfl_xor(rs, 1, 32);
            rs += __shfl_xor(rs, 2, 32);
            rs += __shfl_xor(rs, 4, 32);
            rs += __shfl_xor(rs, 8, 32);
            l_i[r] = l_i[r] * alpha + rs;
            m_i[r] = mnew;
#pragma unroll
            for (int t = 0; t < 8; ++t) acc[t][r] *= alpha;    // rescale O accumulators
            lds_p[(r + 8 * half) * 32 + lr]      = f2bf(p0);   // stage P for transpose
            lds_p[(r + 8 * half) * 32 + 16 + lr] = f2bf(p1);
        }
        __syncthreads();

        // P re-layout into A-fragment: full K=32 (both key halves valid)
        FragAB pf;
        pf.u[0] = *(const u32x4*)(&lds_p[lr * 32 + half * 8]);
        pf.u[1] = *(const u32x4*)(&lds_p[lr * 32 + 16 + half * 8]);

        // O += P * V : 8 output tiles; every lane loads two 8-key runs of Vt (keys 8h+s and 16+8h+s)
#pragma unroll
        for (int t = 0; t < 8; ++t) {
            FragAB vf;
            const __bf16* vp = vbase + (size_t)(16 * t + lr) * S + (size_t)j * 32 + half * 8;
            vf.u[0] = *(const u32x4*)(vp);
            vf.u[1] = *(const u32x4*)(vp + 16);
            acc[t] = __builtin_amdgcn_wmma_f32_16x16x32_bf16(false, pf.v, false, vf.v, (short)0, acc[t], false, false);
        }
        __syncthreads();
    }

    // normalize and emit bf16 [B, S, H*HD] for the output projection
#pragma unroll
    for (int r = 0; r < 8; ++r) {
        float inv = 1.0f / l_i[r];
        int rowg = qt * 16 + r + 8 * half;
        __bf16* ob = O + ((size_t)b * S + rowg) * E + (size_t)h * HD + lr;
#pragma unroll
        for (int t = 0; t < 8; ++t) ob[16 * t] = f2bf(acc[t][r] * inv);
    }
}

// ---------------- host orchestration ----------------
extern "C" void kernel_launch(void* const* d_in, const int* in_sizes, int n_in,
                              void* d_out, int out_size, void* d_ws, size_t ws_size,
                              hipStream_t stream) {
    const float*         x    = (const float*)d_in[0];
    const unsigned char* mask = (const unsigned char*)d_in[1];
    const float*         cosp = (const float*)d_in[2];
    const float*         sinp = (const float*)d_in[3];
    const float*         wq   = (const float*)d_in[4];
    const float*         wk   = (const float*)d_in[5];
    const float*         wv   = (const float*)d_in[6];
    const float*         wo   = (const float*)d_in[7];
    float* out = (float*)d_out;

    char* ws = (char*)d_ws;
    size_t off = 0;
    auto alloc = [&](size_t bytes) -> void* {
        void* p = ws + off;
        off += (bytes + 255) & ~(size_t)255;
        return p;
    };

    const long long NX  = (long long)M * D;
    const long long NWQ = (long long)E * D;
    const long long NWK = (long long)EKV * D;
    const long long NWO = (long long)D * E;

    __bf16* xb  = (__bf16*)alloc(NX  * 2);
    __bf16* wqb = (__bf16*)alloc(NWQ * 2);
    __bf16* wkb = (__bf16*)alloc(NWK * 2);
    __bf16* wvb = (__bf16*)alloc(NWK * 2);
    __bf16* wob = (__bf16*)alloc(NWO * 2);
    float*  qf  = (float*) alloc((size_t)M * E   * 4);
    float*  kf  = (float*) alloc((size_t)M * EKV * 4);
    float*  vf  = (float*) alloc((size_t)M * EKV * 4);
    __bf16* qb  = (__bf16*)alloc((size_t)M * E   * 2);   // [B,H,S,HD] roped
    __bf16* kb  = (__bf16*)alloc((size_t)M * EKV * 2);   // [B,KVH,S,HD] roped
    __bf16* vtb = (__bf16*)alloc((size_t)M * EKV * 2);   // [B,KVH,HD,S]
    __bf16* ao  = (__bf16*)qf;                           // alias: qf dead after RoPE

    // 1) fp32 -> bf16
    cvt_f32_to_bf16<<<4096, 256, 0, stream>>>(x,  xb,  NX);
    cvt_f32_to_bf16<<<4096, 256, 0, stream>>>(wq, wqb, NWQ);
    cvt_f32_to_bf16<<<2048, 256, 0, stream>>>(wk, wkb, NWK);
    cvt_f32_to_bf16<<<2048, 256, 0, stream>>>(wv, wvb, NWK);
    cvt_f32_to_bf16<<<4096, 256, 0, stream>>>(wo, wob, NWO);

    // 2) QKV projections (async-LDS staged WMMA, f32 accumulate)
    wmma_gemm_bf16<<<dim3(E   / 128, M / 64), 128, 0, stream>>>(xb, wqb, qf, M, E,   D);
    wmma_gemm_bf16<<<dim3(EKV / 128, M / 64), 128, 0, stream>>>(xb, wkb, kf, M, EKV, D);
    wmma_gemm_bf16<<<dim3(EKV / 128, M / 64), 128, 0, stream>>>(xb, wvb, vf, M, EKV, D);

    // 3) RoPE + repack to head-major bf16; V transposed for P*V fragment loads
    rope_to_bf16<<<8192, 256, 0, stream>>>(qf, cosp, sinp, qb, H);
    rope_to_bf16<<<2048, 256, 0, stream>>>(kf, cosp, sinp, kb, KVH);
    vtrans_to_bf16<<<4096, 256, 0, stream>>>(vf, vtb);

    // 4) causal GQA flash attention (32-key steps)
    flash_attn<<<dim3(S / 16, H, Bc), 32, 0, stream>>>(qb, kb, vtb, mask, ao);

    // 5) output projection -> fp32 d_out
    wmma_gemm_bf16<<<dim3(D / 128, M / 64), 128, 0, stream>>>(ao, wob, out, M, D, E);
}
